// _LightAttentionLayer_65816078844392
// MI455X (gfx1250) — compile-verified, matrix operand order checked by reading, not run on hardware
//
#include <hip/hip_runtime.h>
#include <hip/hip_bf16.h>
#include <cstdint>

// ---------------------------------------------------------------------------
// MI455X (gfx1250) implementation of the reference transformer layer.
//   B=2, T=2048, D=1024, H=16, HD=64   (all tile sizes divide exactly)
// All matmuls run on v_wmma_f32_16x16x32_bf16 (fp32 -> bf16 inputs, f32
// accumulate). GEMM uses double-buffered LDS with the CDNA5 async
// global->LDS DMA path (global_load_async_to_lds_b128 / s_wait_asynccnt)
// for the pre-packed bf16 weight tiles. Attention is flash-style with online
// softmax. LN/residual/bias/GELU are fused epilogues or cheap kernels.
// ---------------------------------------------------------------------------

typedef __attribute__((ext_vector_type(16))) __bf16 bf16x16;
typedef __attribute__((ext_vector_type(8)))  __bf16 bf16x8;
typedef __attribute__((ext_vector_type(8)))  float  f32x8;

union Frag16 { bf16x16 f; bf16x8 h[2]; };

// Load a 16(row) x 32(K) bf16 fragment in WMMA A/B lane layout from LDS.
// ISA layout (16-bit A 16x32): lane L holds row = L%16;
//   lanes 0-15 : K = {0..7, 16..23},  lanes 16-31 : K = {8..15, 24..31}
// => two 16-byte LDS loads per lane (row stride must keep 16B alignment).
__device__ __forceinline__ bf16x16 load_frag(const __bf16* lds, int rowBase,
                                             int kBase, int stride) {
    int lane  = threadIdx.x & 31;
    int row   = rowBase + (lane & 15);
    int khalf = (lane < 16) ? 0 : 8;
    const __bf16* p = lds + row * stride + kBase + khalf;
    Frag16 fr;
    fr.h[0] = *(const bf16x8*)(p);
    fr.h[1] = *(const bf16x8*)(p + 16);
    return fr.f;
}

__device__ __forceinline__ f32x8 wmma_bf16(bf16x16 a, bf16x16 b, f32x8 c) {
    return __builtin_amdgcn_wmma_f32_16x16x32_bf16(
        /*neg_a=*/false, a, /*neg_b=*/false, b,
        /*c_mod=*/(short)0, c, /*reuse_a=*/false, /*reuse_b=*/false);
}

// Async DMA: copy 16 bytes per lane from global memory straight into LDS,
// tracked by ASYNCcnt (no VGPR round trip).  lds_off is the wave-relative
// LDS byte address (low 32 bits of the flat shared pointer per the CDNA5
// aperture mapping).
__device__ __forceinline__ void async_copy_b128(void* lds_ptr, const void* g) {
    uint32_t lds_off = (uint32_t)(uintptr_t)lds_ptr;
    asm volatile("global_load_async_to_lds_b128 %0, %1, off"
                 :: "v"(lds_off), "v"(g) : "memory");
}

__device__ __forceinline__ void wait_async0() {
    asm volatile("s_wait_asynccnt 0x0" ::: "memory");
}

// ---------------------------------------------------------------------------
// Weight pack: fp32 W[K][N] row-major  ->  bf16 Wt[N][K]  (transposed)
// ---------------------------------------------------------------------------
__global__ void pack_wt(const float* __restrict__ w, __bf16* __restrict__ wt,
                        int K, int N) {
    size_t total = (size_t)K * N;
    for (size_t i = (size_t)blockIdx.x * blockDim.x + threadIdx.x; i < total;
         i += (size_t)gridDim.x * blockDim.x) {
        size_t n = i / K;
        size_t k = i - n * K;
        wt[i] = (__bf16)w[k * (size_t)N + n];
    }
}

// ---------------------------------------------------------------------------
// GEMM:  C[M][N] = act( A[M][K] @ W[K][N] + bias[N] )
//   A     : fp32 row-major (converted to bf16 while staging to LDS)
//   Wt    : bf16, pre-transposed [N][K] (staged via async global->LDS DMA)
// Block tile 128x128x32, 256 threads (8 waves, 4x2), wave tile 32x64
//   => 8 v_wmma_f32_16x16x32_bf16 per wave per K-step.
// Double-buffered LDS: tile t+1 staging (async B DMA + A convert) overlaps
// tile t's WMMAs; single s_wait_asynccnt+barrier per K-step.
// ---------------------------------------------------------------------------
#define BM 128
#define BN 128
#define BK 32
#define LDS_STR 40   // BK + 8 pad; 40*2B = 80B row stride (16B aligned)

template <int ACT>  // 0 = none, 1 = exact GELU
__global__ __launch_bounds__(256) void gemm_bf16w(
    const float* __restrict__ A, const __bf16* __restrict__ Wt,
    const float* __restrict__ bias, float* __restrict__ C,
    int M, int N, int K) {
    __shared__ __bf16 As[2][BM * LDS_STR];
    __shared__ __bf16 Bs[2][BN * LDS_STR];

    const int bm  = blockIdx.y * BM;
    const int bn  = blockIdx.x * BN;
    const int tid = threadIdx.x;
    const int wid = tid >> 5;
    const int wm  = (wid >> 1) * 32;  // 0..96
    const int wn  = (wid & 1) * 64;   // 0,64
    const int lane = tid & 31;

    // Stage one 128x32 K-slice into LDS buffer `buf`.
    auto stage = [&](int k0, int buf) {
        // Weight tile: pure bf16 byte-copy -> async DMA (2 x 16B per thread).
#pragma unroll
        for (int i = 0; i < 2; i++) {
            int idx = tid + i * 256;
            int r   = idx >> 2;          // 4 chunks of 8 bf16 per row
            int c8  = (idx & 3) * 8;
            async_copy_b128(&Bs[buf][r * LDS_STR + c8],
                            Wt + (size_t)(bn + r) * K + k0 + c8);
        }
        // Activation tile: fp32 -> bf16 convert through VGPRs (overlaps DMA).
#pragma unroll
        for (int i = 0; i < 4; i++) {
            int idx = tid + i * 256;
            int r   = idx >> 3;          // 8 float4 per row
            int c4  = (idx & 7) * 4;
            const float* src = A + (size_t)(bm + r) * K + k0 + c4;
            float4 v = *(const float4*)src;
            if (k0 + 2 * BK < K) __builtin_prefetch(src + 2 * BK, 0, 1);
            __bf16* dst = &As[buf][r * LDS_STR + c4];
            dst[0] = (__bf16)v.x; dst[1] = (__bf16)v.y;
            dst[2] = (__bf16)v.z; dst[3] = (__bf16)v.w;
        }
    };

    f32x8 zero = {0.f, 0.f, 0.f, 0.f, 0.f, 0.f, 0.f, 0.f};
    f32x8 acc[2][4];
#pragma unroll
    for (int i = 0; i < 2; i++)
#pragma unroll
        for (int j = 0; j < 4; j++) acc[i][j] = zero;

    stage(0, 0);
    wait_async0();
    __syncthreads();

    const int nk = K / BK;
    for (int t = 0; t < nk; t++) {
        const int buf = t & 1;
        if (t + 1 < nk) stage((t + 1) * BK, buf ^ 1);

        bf16x16 af[2], bfr[4];
        af[0] = load_frag(As[buf], wm,      0, LDS_STR);
        af[1] = load_frag(As[buf], wm + 16, 0, LDS_STR);
#pragma unroll
        for (int j = 0; j < 4; j++)
            bfr[j] = load_frag(Bs[buf], wn + j * 16, 0, LDS_STR);
#pragma unroll
        for (int i = 0; i < 2; i++)
#pragma unroll
            for (int j = 0; j < 4; j++)
                acc[i][j] = wmma_bf16(af[i], bfr[j], acc[i][j]);

        wait_async0();      // next tile's DMA landed in LDS
        __syncthreads();    // next tile's A stores visible; this tile done
    }

    // Epilogue: C layout => VGPR r, lanes 0-15: (row r, col lane),
    // lanes 16-31: (row r+8, col lane-16)
    const int colOff = lane & 15;
    const int rowOff = (lane >> 4) * 8;
#pragma unroll
    for (int i = 0; i < 2; i++) {
#pragma unroll
        for (int j = 0; j < 4; j++) {
            int col  = bn + wn + j * 16 + colOff;
            float bv = bias[col];
#pragma unroll
            for (int r = 0; r < 8; r++) {
                int row   = bm + wm + i * 16 + rowOff + r;
                float val = acc[i][j][r] + bv;
                if (ACT == 1)
                    val = 0.5f * val * (1.0f + erff(val * 0.70710678118654752f));
                C[(size_t)row * N + col] = val;
            }
        }
    }
}

// ---------------------------------------------------------------------------
// Flash attention (shared Q/K projection): out = softmax(QK^T/8) V
//   qk,v : [B, T, H*HD] fp32      out : [B, T, H*HD] fp32
// grid (T/64, H, B), 128 threads (4 waves). Each wave owns a 16-query strip.
// ---------------------------------------------------------------------------
#define AHD 64
#define ASTR 72   // 64 + 8 pad, 144B row stride (16B aligned)

__global__ __launch_bounds__(128) void attn_kernel(
    const float* __restrict__ qk, const float* __restrict__ v,
    float* __restrict__ out, int B, int T, int H) {
    __shared__ __bf16 Qs[64 * ASTR];
    __shared__ __bf16 Ks[64 * ASTR];
    __shared__ __bf16 Vt[64 * ASTR];   // transposed: [hd][key]
    __shared__ __bf16 Ps[64 * ASTR];

    const int b  = blockIdx.z;
    const int h  = blockIdx.y;
    const int q0 = blockIdx.x * 64;
    const int tid = threadIdx.x;
    const int wid = tid >> 5;
    const int lane = tid & 31;
    const size_t Dm = (size_t)H * AHD;

    const float* qbase = qk + (size_t)b * T * Dm + (size_t)h * AHD;
    const float* vbase = v  + (size_t)b * T * Dm + (size_t)h * AHD;

    // Stage Q (64x64 fp32 -> bf16): 1024 float4, 8 per thread.
#pragma unroll
    for (int i = 0; i < 8; i++) {
        int idx = tid + i * 128;
        int r   = idx >> 4;
        int c4  = (idx & 15) * 4;
        float4 qv = *(const float4*)(qbase + (size_t)(q0 + r) * Dm + c4);
        __bf16* dst = Qs + r * ASTR + c4;
        dst[0] = (__bf16)qv.x; dst[1] = (__bf16)qv.y;
        dst[2] = (__bf16)qv.z; dst[3] = (__bf16)qv.w;
    }
    __syncthreads();

    bf16x16 qa[2];
    qa[0] = load_frag(Qs, wid * 16, 0,  ASTR);
    qa[1] = load_frag(Qs, wid * 16, 32, ASTR);

    f32x8 zero = {0.f, 0.f, 0.f, 0.f, 0.f, 0.f, 0.f, 0.f};
    f32x8 o[4];
#pragma unroll
    for (int n = 0; n < 4; n++) o[n] = zero;
    float m_r[8], l_r[8];
#pragma unroll
    for (int r = 0; r < 8; r++) { m_r[r] = -1e30f; l_r[r] = 0.f; }

    const int colOff = lane & 15;
    const int rowOff = (lane >> 4) * 8;

    for (int kb = 0; kb < T; kb += 64) {
        __syncthreads();
        // Stage K tile and transposed V tile.
#pragma unroll
        for (int i = 0; i < 8; i++) {
            int idx = tid + i * 128;
            int r   = idx >> 4;
            int c4  = (idx & 15) * 4;
            float4 kv = *(const float4*)(qbase + (size_t)(kb + r) * Dm + c4);
            __bf16* dst = Ks + r * ASTR + c4;
            dst[0] = (__bf16)kv.x; dst[1] = (__bf16)kv.y;
            dst[2] = (__bf16)kv.z; dst[3] = (__bf16)kv.w;
            float4 vv = *(const float4*)(vbase + (size_t)(kb + r) * Dm + c4);
            Vt[(c4 + 0) * ASTR + r] = (__bf16)vv.x;
            Vt[(c4 + 1) * ASTR + r] = (__bf16)vv.y;
            Vt[(c4 + 2) * ASTR + r] = (__bf16)vv.z;
            Vt[(c4 + 3) * ASTR + r] = (__bf16)vv.w;
        }
        __syncthreads();

        // S = Q K^T (16 queries x 64 keys per wave), K-dim = HD = 64.
        f32x8 s[4];
#pragma unroll
        for (int n = 0; n < 4; n++) {
            bf16x16 kb0 = load_frag(Ks, n * 16, 0,  ASTR);
            bf16x16 kb1 = load_frag(Ks, n * 16, 32, ASTR);
            s[n] = wmma_bf16(qa[0], kb0, zero);
            s[n] = wmma_bf16(qa[1], kb1, s[n]);
        }
        // scale by 1/sqrt(HD) = 1/8
#pragma unroll
        for (int n = 0; n < 4; n++)
#pragma unroll
            for (int r = 0; r < 8; r++) s[n][r] *= 0.125f;

        // Online softmax per output row (xor masks 1..8 stay in 16-lane group)
#pragma unroll
        for (int r = 0; r < 8; r++) {
            float mx = fmaxf(fmaxf(s[0][r], s[1][r]), fmaxf(s[2][r], s[3][r]));
#pragma unroll
            for (int off = 8; off > 0; off >>= 1)
                mx = fmaxf(mx, __shfl_xor(mx, off));
            float mnew = fmaxf(m_r[r], mx);
            float corr = __expf(m_r[r] - mnew);
            float psum = 0.f;
#pragma unroll
            for (int n = 0; n < 4; n++) {
                float p = __expf(s[n][r] - mnew);
                psum += p;
                Ps[(wid * 16 + rowOff + r) * ASTR + n * 16 + colOff] = (__bf16)p;
            }
#pragma unroll
            for (int off = 8; off > 0; off >>= 1)
                psum += __shfl_xor(psum, off);
            l_r[r] = l_r[r] * corr + psum;
            m_r[r] = mnew;
#pragma unroll
            for (int n = 0; n < 4; n++) o[n][r] *= corr;
        }

        // O += P V   (K-dim = 64 keys, two 32-steps). Ps is wave-private.
        bf16x16 pa0 = load_frag(Ps, wid * 16, 0,  ASTR);
        bf16x16 pa1 = load_frag(Ps, wid * 16, 32, ASTR);
#pragma unroll
        for (int n = 0; n < 4; n++) {
            bf16x16 vb0 = load_frag(Vt, n * 16, 0,  ASTR);
            bf16x16 vb1 = load_frag(Vt, n * 16, 32, ASTR);
            o[n] = wmma_bf16(pa0, vb0, o[n]);
            o[n] = wmma_bf16(pa1, vb1, o[n]);
        }
    }

    // Normalize and write out.
#pragma unroll
    for (int r = 0; r < 8; r++) {
        float inv = 1.0f / l_r[r];
        int row = q0 + wid * 16 + rowOff + r;
#pragma unroll
        for (int n = 0; n < 4; n++) {
            out[((size_t)b * T + row) * Dm + (size_t)h * AHD + n * 16 + colOff] =
                o[n][r] * inv;
        }
    }
}

// ---------------------------------------------------------------------------
// Fused residual + LayerNorm: out = LN(x + res) * g + b.  One block per token.
// ---------------------------------------------------------------------------
__global__ __launch_bounds__(256) void ln_kernel(
    const float* __restrict__ x, const float* __restrict__ res,
    const float* __restrict__ g, const float* __restrict__ be,
    float* __restrict__ out, int D) {
    __shared__ float red[2][8];
    const size_t base = (size_t)blockIdx.x * D;
    float v[4], s = 0.f, s2 = 0.f;
#pragma unroll
    for (int i = 0; i < 4; i++) {
        int c = threadIdx.x + i * 256;
        v[i] = x[base + c] + res[base + c];
        s += v[i];
        s2 += v[i] * v[i];
    }
#pragma unroll
    for (int off = 16; off > 0; off >>= 1) {
        s  += __shfl_xor(s, off);
        s2 += __shfl_xor(s2, off);
    }
    if ((threadIdx.x & 31) == 0) {
        red[0][threadIdx.x >> 5] = s;
        red[1][threadIdx.x >> 5] = s2;
    }
    __syncthreads();
    float ts = 0.f, ts2 = 0.f;
#pragma unroll
    for (int i = 0; i < 8; i++) { ts += red[0][i]; ts2 += red[1][i]; }
    float mean = ts / (float)D;
    float var  = ts2 / (float)D - mean * mean;
    float rstd = rsqrtf(var + 1e-5f);
#pragma unroll
    for (int i = 0; i < 4; i++) {
        int c = threadIdx.x + i * 256;
        out[base + c] = (v[i] - mean) * rstd * g[c] + be[c];
    }
}

// ---------------------------------------------------------------------------
// Orchestration
// ---------------------------------------------------------------------------
extern "C" void kernel_launch(void* const* d_in, const int* in_sizes, int n_in,
                              void* d_out, int out_size, void* d_ws,
                              size_t ws_size, hipStream_t stream) {
    const int B = 2, T = 2048, D = 1024, H = 16;
    const int M = B * T;  // 4096 tokens

    const float* x    = (const float*)d_in[0];
    const float* w_qk = (const float*)d_in[1];
    const float* b_qk = (const float*)d_in[2];
    const float* w_v  = (const float*)d_in[3];
    const float* b_v  = (const float*)d_in[4];
    const float* w_o  = (const float*)d_in[5];
    const float* b_o  = (const float*)d_in[6];
    const float* g1   = (const float*)d_in[7];
    const float* be1  = (const float*)d_in[8];
    const float* g2   = (const float*)d_in[9];
    const float* be2  = (const float*)d_in[10];
    const float* w_f1 = (const float*)d_in[11];
    const float* b_f1 = (const float*)d_in[12];
    const float* w_f2 = (const float*)d_in[13];
    const float* b_f2 = (const float*)d_in[14];
    float* outp = (float*)d_out;

    // Workspace layout (fp32 activations then bf16 packed weights), ~115 MB.
    float* ws = (float*)d_ws;
    const size_t SZ = (size_t)M * D;          // 4M elements per activation
    float* qk_buf = ws;                       // qk; reused as o-proj output
    float* v_buf  = ws + SZ;                  // v;  reused as ffn output
    float* a_buf  = ws + 2 * SZ;              // attention output
    float* x1_buf = ws + 3 * SZ;              // post-LN1
    float* h_buf  = ws + 4 * SZ;              // GELU hidden (2*SZ elements)
    __bf16* wt_qk = (__bf16*)(ws + 6 * SZ);
    __bf16* wt_v  = wt_qk + (size_t)D * D;
    __bf16* wt_o  = wt_v  + (size_t)D * D;
    __bf16* wt_f1 = wt_o  + (size_t)D * D;    // [2D][D]
    __bf16* wt_f2 = wt_f1 + (size_t)2 * D * D; // [D][2D]

    // 1) Pack weights to bf16 transposed.
    pack_wt<<<2048, 256, 0, stream>>>(w_qk, wt_qk, D, D);
    pack_wt<<<2048, 256, 0, stream>>>(w_v,  wt_v,  D, D);
    pack_wt<<<2048, 256, 0, stream>>>(w_o,  wt_o,  D, D);
    pack_wt<<<2048, 256, 0, stream>>>(w_f1, wt_f1, D, 2 * D);
    pack_wt<<<2048, 256, 0, stream>>>(w_f2, wt_f2, 2 * D, D);

    dim3 gNN(D / BN, M / BM);       // N = 1024
    dim3 gN2(2 * D / BN, M / BM);   // N = 2048

    // 2) Projections.
    gemm_bf16w<0><<<gNN, 256, 0, stream>>>(x, wt_qk, b_qk, qk_buf, M, D, D);
    gemm_bf16w<0><<<gNN, 256, 0, stream>>>(x, wt_v,  b_v,  v_buf,  M, D, D);

    // 3) Attention.
    attn_kernel<<<dim3(T / 64, H, B), 128, 0, stream>>>(qk_buf, v_buf, a_buf,
                                                        B, T, H);

    // 4) Output projection (reuse qk_buf).
    gemm_bf16w<0><<<gNN, 256, 0, stream>>>(a_buf, wt_o, b_o, qk_buf, M, D, D);

    // 5) Residual + LN1.
    ln_kernel<<<M, 256, 0, stream>>>(x, qk_buf, g1, be1, x1_buf, D);

    // 6) FFN.
    gemm_bf16w<1><<<gN2, 256, 0, stream>>>(x1_buf, wt_f1, b_f1, h_buf,
                                           M, 2 * D, D);
    gemm_bf16w<0><<<gNN, 256, 0, stream>>>(h_buf, wt_f2, b_f2, v_buf,
                                           M, D, 2 * D);

    // 7) Residual + LN2 -> d_out.
    ln_kernel<<<M, 256, 0, stream>>>(x1_buf, v_buf, g2, be2, outp, D);
}
